// Attention_40699110097471
// MI455X (gfx1250) — matrix operand order repaired and checked by reading, not compiled, not run on hardware
//
#include <hip/hip_runtime.h>
#include <hip/hip_bf16.h>

// ---- CDNA5 (gfx1250) flash attention, bf16 WMMA + async-LDS staging ----
// Shapes: B=4, T=4096, C=768, H=64. wave32 only.

typedef __bf16 bf16_t;
typedef __attribute__((ext_vector_type(16))) __bf16 v16bf;
typedef __attribute__((ext_vector_type(8)))  __bf16 v8bf;
typedef __attribute__((ext_vector_type(8)))  float  v8f;
typedef __attribute__((ext_vector_type(4)))  int    v4i;

typedef __attribute__((address_space(1))) v4i as1_v4i;
typedef __attribute__((address_space(3))) v4i as3_v4i;

#define TSEQ   4096
#define NB     4
#define CMODEL 768
#define HSZ    64
#define CHUNK  64          // keys per staged chunk

union V16 { v16bf v; v8bf h[2]; };

#if defined(__has_builtin)
#  if __has_builtin(__builtin_amdgcn_global_load_async_to_lds_b128) && \
      __has_builtin(__builtin_amdgcn_s_wait_asynccnt)
#    define USE_ASYNC_LDS 1
#  endif
#endif
#ifndef USE_ASYNC_LDS
#  define USE_ASYNC_LDS 0
#endif

__device__ __forceinline__ v8bf ldg8(const bf16_t* p) { return *(const v8bf*)p; }
__device__ __forceinline__ v8f  wmma_bf16(const V16& a, const V16& b, v8f c) {
    return __builtin_amdgcn_wmma_f32_16x16x32_bf16(false, a.v, false, b.v,
                                                   (short)0, c, false, false);
}
__device__ __forceinline__ float hmax16(float x) {
    #pragma unroll
    for (int m = 1; m < 16; m <<= 1) x = fmaxf(x, __shfl_xor(x, m, 32));
    return x;
}
__device__ __forceinline__ float hsum16(float x) {
    #pragma unroll
    for (int m = 1; m < 16; m <<= 1) x += __shfl_xor(x, m, 32);
    return x;
}
__device__ __forceinline__ v8bf cvt8(v8f f) {
    v8bf r;
    #pragma unroll
    for (int i = 0; i < 8; ++i) r[i] = (bf16_t)f[i];
    return r;
}

// Copy 16 bytes global -> LDS (async if available on this toolchain).
__device__ __forceinline__ void async_cp16(const bf16_t* g, bf16_t* l) {
#if USE_ASYNC_LDS
    __builtin_amdgcn_global_load_async_to_lds_b128(
        (as1_v4i*)const_cast<bf16_t*>(g), (as3_v4i*)l, 0, 0);
#else
    *(v8bf*)l = *(const v8bf*)g;
#endif
}
__device__ __forceinline__ void async_wait_all() {
#if USE_ASYNC_LDS
    __builtin_amdgcn_s_wait_asynccnt(0);
#endif
}

// ---------- Kernel 0: W [C,H] f32 -> Wt [H,C] bf16, for q/k/v ----------
__global__ void wtrans_kernel(const float* __restrict__ Wq,
                              const float* __restrict__ Wk,
                              const float* __restrict__ Wv,
                              bf16_t* __restrict__ wt) {
    int idx = blockIdx.x * blockDim.x + threadIdx.x;
    if (idx >= 3 * HSZ * CMODEL) return;
    int w   = idx / (HSZ * CMODEL);
    int rem = idx % (HSZ * CMODEL);
    int h   = rem / CMODEL;
    int c   = rem % CMODEL;
    const float* W = (w == 0) ? Wq : (w == 1) ? Wk : Wv;
    wt[idx] = (bf16_t)W[c * HSZ + h];
}

// ---------- Kernel 1: projections q,k (row-major bf16) and v^T ----------
__global__ __launch_bounds__(32) void proj_kernel(
        const float* __restrict__ x, const bf16_t* __restrict__ wt,
        bf16_t* __restrict__ qb, bf16_t* __restrict__ kb,
        bf16_t* __restrict__ vt) {
    const int lane = threadIdx.x;
    const int lm   = lane & 15;
    const int half = lane >> 4;
    const int g0   = blockIdx.x * 16;          // global row tile in [0, B*T)

    v8f acc[3][4];
    #pragma unroll
    for (int w = 0; w < 3; ++w)
        #pragma unroll
        for (int nf = 0; nf < 4; ++nf) acc[w][nf] = (v8f)(0.0f);

    const float* xr = x + (size_t)(g0 + lm) * CMODEL + half * 8;

    for (int kc = 0; kc < CMODEL; kc += 32) {
        // A fragment: row = g0+lm, features kc + half*8 + {0..7, 16..23}
        V16 a;
        {
            const float* p0 = xr + kc;
            v8f f0 = *(const v8f*)(p0);
            v8f f1 = *(const v8f*)(p0 + 16);
            a.h[0] = cvt8(f0);
            a.h[1] = cvt8(f1);
        }
        #pragma unroll
        for (int w = 0; w < 3; ++w) {
            #pragma unroll
            for (int nf = 0; nf < 4; ++nf) {
                // B fragment: N = nf*16 + lm, K = kc + half*8 + {0..7, 16..23}
                const bf16_t* wrow = wt + ((size_t)w * HSZ + nf * 16 + lm) * CMODEL
                                        + kc + half * 8;
                V16 b;
                b.h[0] = ldg8(wrow);
                b.h[1] = ldg8(wrow + 16);
                acc[w][nf] = wmma_bf16(a, b, acc[w][nf]);
            }
        }
    }

    // C/D layout: row m = r + 8*half, col n = lm
    #pragma unroll
    for (int nf = 0; nf < 4; ++nf) {
        #pragma unroll
        for (int r = 0; r < 8; ++r) {
            int m = g0 + r + 8 * half;
            int h = nf * 16 + lm;
            qb[(size_t)m * HSZ + h] = (bf16_t)acc[0][nf][r];
            kb[(size_t)m * HSZ + h] = (bf16_t)acc[1][nf][r];
            int b = m >> 12;                 // T = 4096
            int t = m & (TSEQ - 1);
            vt[((size_t)b * HSZ + h) * TSEQ + t] = (bf16_t)acc[2][nf][r];
        }
    }
}

// ---------- Kernel 2: causal flash attention ----------
// 128-thread block = 4 waves = 64 query rows. 64-key chunks of K/V staged
// cooperatively into double-buffered LDS with async loads; each wave runs
// online softmax + WMMAs on its own 16-row tile. Wave-uniform causal skip.
__global__ __launch_bounds__(128) void attn_kernel(
        const bf16_t* __restrict__ qb, const bf16_t* __restrict__ kb,
        const bf16_t* __restrict__ vt, float* __restrict__ out) {
    __shared__ __align__(16) bf16_t kbuf[2][CHUNK * HSZ];   // [key][feat]  8KB each
    __shared__ __align__(16) bf16_t vbuf[2][HSZ * CHUNK];   // [feat][key]  8KB each
    __shared__ __align__(16) bf16_t pbuf[4][16 * CHUNK];    // per-wave P    2KB each

    const int tid  = threadIdx.x;
    const int wid  = tid >> 5;
    const int lane = tid & 31;
    const int lm   = lane & 15;
    const int half = lane >> 4;
    const int b    = blockIdx.y;
    const int blkM0 = blockIdx.x * 64;
    const int m0    = blkM0 + wid * 16;          // this wave's query tile
    const size_t rowbase = (size_t)b * TSEQ;

    // Q A-fragments (feature chunks 0..31 / 32..63)
    V16 aq0, aq1;
    {
        const bf16_t* qr = qb + (rowbase + m0 + lm) * HSZ + half * 8;
        aq0.h[0] = ldg8(qr);       aq0.h[1] = ldg8(qr + 16);
        aq1.h[0] = ldg8(qr + 32);  aq1.h[1] = ldg8(qr + 48);
    }

    float rmax[8], lsum[8];
    v8f acc[4];
    #pragma unroll
    for (int r = 0; r < 8; ++r) { rmax[r] = -1e30f; lsum[r] = 0.0f; }
    #pragma unroll
    for (int nf = 0; nf < 4; ++nf) acc[nf] = (v8f)(0.0f);

    const int nch_blk = (blkM0 + 127) >> 6;      // chunks needed by last wave
    const int nch_w   = (m0 + 79) >> 6;          // chunks this wave computes

    // --- cooperative stage of one 64-key chunk into buffer `buf` ---
    auto stage = [&](int s0, int buf) {
        // K: 64 rows x 128B contiguous (8KB); each thread copies 64B
        {
            const bf16_t* src = kb + (rowbase + s0) * HSZ + tid * 32;
            bf16_t*       dst = kbuf[buf] + tid * 32;
            #pragma unroll
            for (int i = 0; i < 4; ++i) async_cp16(src + i * 8, dst + i * 8);
        }
        // V^T: 64 feature rows x 128B, row stride T*2B; thread -> half a row
        {
            int feat = tid >> 1, part = tid & 1;
            const bf16_t* src = vt + ((size_t)b * HSZ + feat) * TSEQ + s0 + part * 32;
            bf16_t*       dst = vbuf[buf] + feat * CHUNK + part * 32;
            #pragma unroll
            for (int i = 0; i < 4; ++i) async_cp16(src + i * 8, dst + i * 8);
        }
    };

    stage(0, 0);                                  // prologue fill

    for (int ch = 0; ch < nch_blk; ++ch) {
        async_wait_all();
        __syncthreads();                          // buffer ch&1 is ready
        if (ch + 1 < nch_blk) stage((ch + 1) << 6, (ch + 1) & 1);

        if (ch < nch_w) {                         // wave-uniform causal skip
            const int s0 = ch << 6;
            const bf16_t* ksh = kbuf[ch & 1];
            const bf16_t* vsh = vbuf[ch & 1];
            bf16_t*       pw  = pbuf[wid];        // wave-private: no barrier

            // ---- scores: 4 key tiles x 2 feature chunks ----
            v8f st[4];
            #pragma unroll
            for (int kt = 0; kt < 4; ++kt) {
                const bf16_t* kr = ksh + (kt * 16 + lm) * HSZ + half * 8;
                V16 bk0, bk1;
                bk0.h[0] = *(const v8bf*)(kr);      bk0.h[1] = *(const v8bf*)(kr + 16);
                bk1.h[0] = *(const v8bf*)(kr + 32); bk1.h[1] = *(const v8bf*)(kr + 48);
                v8f s = (v8f)(0.0f);
                s = wmma_bf16(aq0, bk0, s);
                s = wmma_bf16(aq1, bk1, s);
                st[kt] = s;
            }

            const bool need_mask = (s0 + 63 > m0);
            float alpha[8];
            #pragma unroll
            for (int r = 0; r < 8; ++r) {
                const int m  = m0 + r + 8 * half;
                const int rl = r + 8 * half;
                float p[4];
                float cm = -1e30f;
                #pragma unroll
                for (int kt = 0; kt < 4; ++kt) {
                    float a = st[kt][r] * 0.125f;           // 1/sqrt(64)
                    if (need_mask && (s0 + kt * 16 + lm > m)) a = -1e30f;
                    p[kt] = a;
                    cm = fmaxf(cm, a);
                }
                cm = hmax16(cm);
                float nm = fmaxf(rmax[r], cm);
                float al = __expf(rmax[r] - nm);
                rmax[r]  = nm;
                alpha[r] = al;
                float ps = 0.0f;
                #pragma unroll
                for (int kt = 0; kt < 4; ++kt) {
                    float e = __expf(p[kt] - nm);
                    ps += e;
                    pw[rl * CHUNK + kt * 16 + lm] = (bf16_t)e;
                }
                lsum[r] = lsum[r] * al + hsum16(ps);
            }

            // ---- re-stripe P into A layout (same-wave LDS ops are in-order)
            V16 ap0, ap1;
            {
                const bf16_t* pr = pw + lm * CHUNK + half * 8;
                ap0.h[0] = *(const v8bf*)(pr);      ap0.h[1] = *(const v8bf*)(pr + 16);
                ap1.h[0] = *(const v8bf*)(pr + 32); ap1.h[1] = *(const v8bf*)(pr + 48);
            }

            // ---- PV: 4 feature tiles, K = 64 keys (2 WMMAs each) ----
            #pragma unroll
            for (int nf = 0; nf < 4; ++nf) {
                const bf16_t* vr = vsh + (nf * 16 + lm) * CHUNK + half * 8;
                V16 bv0, bv1;
                bv0.h[0] = *(const v8bf*)(vr);      bv0.h[1] = *(const v8bf*)(vr + 16);
                bv1.h[0] = *(const v8bf*)(vr + 32); bv1.h[1] = *(const v8bf*)(vr + 48);
                v8f c = acc[nf];
                #pragma unroll
                for (int r = 0; r < 8; ++r) c[r] *= alpha[r];
                c = wmma_bf16(ap0, bv0, c);
                acc[nf] = wmma_bf16(ap1, bv1, c);
            }
        }
    }

    // ---- epilogue: out[b, m, h] = acc / lsum ----
    float inv[8];
    #pragma unroll
    for (int r = 0; r < 8; ++r) inv[r] = 1.0f / lsum[r];
    #pragma unroll
    for (int nf = 0; nf < 4; ++nf) {
        #pragma unroll
        for (int r = 0; r < 8; ++r) {
            int m = m0 + r + 8 * half;
            out[(rowbase + m) * HSZ + nf * 16 + lm] = acc[nf][r] * inv[r];
        }
    }
}

extern "C" void kernel_launch(void* const* d_in, const int* in_sizes, int n_in,
                              void* d_out, int out_size, void* d_ws, size_t ws_size,
                              hipStream_t stream) {
    const float* x  = (const float*)d_in[0];
    const float* Wq = (const float*)d_in[1];
    const float* Wk = (const float*)d_in[2];
    const float* Wv = (const float*)d_in[3];
    float* out = (float*)d_out;

    // Workspace (bf16): q[B*T,64], k[B*T,64], vT[B,64,T], Wt[3,64,768] ~6.6MB
    const size_t NROWS = (size_t)NB * TSEQ;           // 16384
    bf16_t* qb = (bf16_t*)d_ws;
    bf16_t* kb = qb + NROWS * HSZ;
    bf16_t* vt = kb + NROWS * HSZ;
    bf16_t* wt = vt + NROWS * HSZ;

    int wtn = 3 * HSZ * CMODEL;
    wtrans_kernel<<<(wtn + 255) / 256, 256, 0, stream>>>(Wq, Wk, Wv, wt);

    proj_kernel<<<(int)(NROWS / 16), 32, 0, stream>>>(x, wt, qb, kb, vt);

    dim3 grid(TSEQ / 64, NB);
    attn_kernel<<<grid, 128, 0, stream>>>(qb, kb, vt, out);
}
